// Fourdloss_16621523436340
// MI455X (gfx1250) — compile-verified, matrix operand order checked by reading, not run on hardware
//
#include <hip/hip_runtime.h>

typedef __attribute__((ext_vector_type(16))) __bf16 v16bf;
typedef __attribute__((ext_vector_type(8)))  float  v8f;
typedef __attribute__((ext_vector_type(4)))  int    i32x4;
typedef __attribute__((ext_vector_type(4)))  unsigned int u32x4;

#define TPB 256
#define USE_ASYNC_STAGE 1

// ---------------------------------------------------------------------------
// Stage 0: bf16 hi/lo split of cf (view-major reshape of features) stored in
// WMMA-fragment-major order, plus expanded labels.
// Fragment layout (16-bit A-matrix 16x32, wave32, ISA 7.12.2):
//   frag(t, kc, lane, e):  row = t*16 + (lane&15); half = lane>>4; v = e>>1;
//   K = kc*32 + 2*((v&3) + 4*half + 8*(v>>2)) + (e&1)
// ---------------------------------------------------------------------------
__global__ __launch_bounds__(TPB) void fourd_prep(
    const float* __restrict__ feat, const int* __restrict__ labels,
    __bf16* __restrict__ Ahi, __bf16* __restrict__ Alo,
    int* __restrict__ labx, int N, int d, int bsz, int kchunks)
{
  long id = (long)blockIdx.x * TPB + threadIdx.x;
  if (id < N) labx[(int)id] = labels[(int)(id % bsz)];
  long total = (long)N * d;
  if (id >= total) return;

  int  e    = (int)(id & 15);
  long r1   = id >> 4;
  int  lane = (int)(r1 & 31);
  long r2   = r1 >> 5;
  int  kc   = (int)(r2 % kchunks);
  int  t    = (int)(r2 / kchunks);

  int h = lane >> 4;
  int m = lane & 15;
  int v = e >> 1;
  int K = kc * 32 + 2 * ((v & 3) + 4 * h + 8 * (v >> 2)) + (e & 1);
  int rg = t * 16 + m;                       // global row of cf
  float x = feat[(size_t)(rg % bsz) * (2 * (size_t)d) + (size_t)(rg / bsz) * d + K];
  __bf16 hi = (__bf16)x;
  float  hf = (float)hi;
  __bf16 lo = (__bf16)(x - hf);
  Ahi[id] = hi;
  Alo[id] = lo;
}

// ---------------------------------------------------------------------------
// Online masked logsumexp update for one 8-value accumulator slice.
// ---------------------------------------------------------------------------
__device__ __forceinline__ void online_update(
    float& m_run, float& s_run, const v8f& acc,
    const i32x4& lj0, const i32x4& lj1, int lab_i, float invT)
{
  float tmax = -3.0e38f;
  float lg[8];
  int   vl[8];
  #pragma unroll
  for (int r = 0; r < 8; ++r) {
    int labj = (r < 4) ? lj0[r] : lj1[r - 4];
    float x = acc[r] * invT;
    lg[r] = x;
    vl[r] = (labj != lab_i);
    if (vl[r]) tmax = fmaxf(tmax, x);
  }
  float m_new = fmaxf(m_run, tmax);
  float ss = 0.0f;
  #pragma unroll
  for (int r = 0; r < 8; ++r)
    ss += vl[r] ? __expf(lg[r] - m_new) : 0.0f;
  s_run = s_run * __expf(m_run - m_new) + ss;
  m_run = m_new;
}

// ---------------------------------------------------------------------------
// Stage 1: fused bf16x3 GEMM + masked online logsumexp.
// Block = 8 waves = 8 column groups; every wave processes BOTH of the block's
// two 16-anchor row-tiles per A-frag load (6 WMMAs per hi/lo A pair),
// halving L2 traffic vs one-tile-per-wave. B operand (anchor rows) staged in
// LDS via gfx1250 async loads; A operand streamed from L2 (cf = 16MB).
// ---------------------------------------------------------------------------
__global__ __launch_bounds__(TPB) void fourd_main(
    const __bf16* __restrict__ Ahi, const __bf16* __restrict__ Alo,
    const int* __restrict__ labx, float* __restrict__ S,
    int N, int kchunks, int ctPerGroup)
{
  __shared__ v16bf smh[1024];          // 2 tiles * 16 chunks * 32 lanes (32KB)
  __shared__ v16bf sml[1024];
  __shared__ float pm[2][8][16];
  __shared__ float ps[2][8][16];

  const v16bf* Gh = (const v16bf*)Ahi;
  const v16bf* Gl = (const v16bf*)Alo;
  int nf = 2 * kchunks * 32;           // v16bf frags to stage per array
  int gbase = blockIdx.x * nf;

#if USE_ASYNC_STAGE
  {
    const u32x4* gsrcH = (const u32x4*)(Gh + gbase);
    const u32x4* gsrcL = (const u32x4*)(Gl + gbase);
    unsigned ldsH = (unsigned)(size_t)(void*)&smh[0];  // low 32 bits = LDS addr
    unsigned ldsL = (unsigned)(size_t)(void*)&sml[0];
    int n16 = nf * 2;                  // b128 transfers per array
    for (int idx = threadIdx.x; idx < n16; idx += TPB) {
      unsigned dH = ldsH + (unsigned)idx * 16u;
      unsigned dL = ldsL + (unsigned)idx * 16u;
      const u32x4* pH = gsrcH + idx;
      const u32x4* pL = gsrcL + idx;
      asm volatile("global_load_async_to_lds_b128 %0, %1, off"
                   :: "v"(dH), "v"(pH) : "memory");
      asm volatile("global_load_async_to_lds_b128 %0, %1, off"
                   :: "v"(dL), "v"(pL) : "memory");
    }
    asm volatile("s_wait_asynccnt 0x0" ::: "memory");
  }
#else
  for (int idx = threadIdx.x; idx < nf; idx += TPB) {
    smh[idx] = Gh[gbase + idx];
    sml[idx] = Gl[gbase + idx];
  }
#endif
  __syncthreads();

  int w    = threadIdx.x >> 5;         // column group (0..7)
  int lane = threadIdx.x & 31;
  int h    = lane >> 4;
  int ln   = lane & 15;
  int lab_i0 = labx[(blockIdx.x * 2 + 0) * 16 + ln];
  int lab_i1 = labx[(blockIdx.x * 2 + 1) * 16 + ln];

  float m0 = -3.0e38f, s0 = 0.0f;
  float m1 = -3.0e38f, s1 = 0.0f;
  const float invT = 1.0f / 0.035f;

  int ct0 = w * ctPerGroup;
  int ctEnd = ct0 + ctPerGroup;
  for (int jt = ct0; jt < ctEnd; ++jt) {
    i32x4 lj0 = *(const i32x4*)(labx + jt * 16 + 8 * h);
    i32x4 lj1 = *(const i32x4*)(labx + jt * 16 + 8 * h + 4);

    v8f acc0 = {}, acc1 = {};
    int gfi  = jt * kchunks * 32 + lane;   // global fragment index
    int sfi0 = lane;                       // LDS frag index, row-tile 0
    int sfi1 = kchunks * 32 + lane;        // LDS frag index, row-tile 1
    if (jt + 1 < ctEnd)
      __builtin_prefetch((const void*)&Gh[gfi + kchunks * 32], 0, 1);
    for (int kc = 0; kc < kchunks; ++kc) {
      v16bf ah = Gh[gfi];
      v16bf al = Gl[gfi];
      v16bf bh0 = smh[sfi0];
      v16bf bl0 = sml[sfi0];
      v16bf bh1 = smh[sfi1];
      v16bf bl1 = sml[sfi1];
      acc0 = __builtin_amdgcn_wmma_f32_16x16x32_bf16(false, ah, false, bh0, (short)0, acc0, false, false);
      acc0 = __builtin_amdgcn_wmma_f32_16x16x32_bf16(false, ah, false, bl0, (short)0, acc0, false, false);
      acc0 = __builtin_amdgcn_wmma_f32_16x16x32_bf16(false, al, false, bh0, (short)0, acc0, false, false);
      acc1 = __builtin_amdgcn_wmma_f32_16x16x32_bf16(false, ah, false, bh1, (short)0, acc1, false, false);
      acc1 = __builtin_amdgcn_wmma_f32_16x16x32_bf16(false, ah, false, bl1, (short)0, acc1, false, false);
      acc1 = __builtin_amdgcn_wmma_f32_16x16x32_bf16(false, al, false, bh1, (short)0, acc1, false, false);
      gfi  += 32;
      sfi0 += 32;
      sfi1 += 32;
    }

    online_update(m0, s0, acc0, lj0, lj1, lab_i0, invT);
    online_update(m1, s1, acc1, lj0, lj1, lab_i1, invT);
  }

  // merge the two half-lanes sharing the same anchor row (per row-tile)
  float mo0 = __shfl_xor(m0, 16, 32);
  float so0 = __shfl_xor(s0, 16, 32);
  float M0 = fmaxf(m0, mo0);
  float S0 = s0 * __expf(m0 - M0) + so0 * __expf(mo0 - M0);

  float mo1 = __shfl_xor(m1, 16, 32);
  float so1 = __shfl_xor(s1, 16, 32);
  float M1 = fmaxf(m1, mo1);
  float S1 = s1 * __expf(m1 - M1) + so1 * __expf(mo1 - M1);

  if (lane < 16) {
    pm[0][w][lane] = M0; ps[0][w][lane] = S0;
    pm[1][w][lane] = M1; ps[1][w][lane] = S1;
  }
  __syncthreads();

  // merge the 8 column groups per row tile, emit S[i]
  if (w < 2 && lane < 16) {
    float M = -3.0e38f, Sm = 0.0f;
    #pragma unroll
    for (int q = 0; q < 8; ++q) {
      float mq = pm[w][q][lane];
      float sq = ps[w][q][lane];
      float nM = fmaxf(M, mq);
      Sm = Sm * __expf(M - nM) + sq * __expf(mq - nM);
      M = nM;
    }
    S[(blockIdx.x * 2 + w) * 16 + lane] = Sm;
  }
}

// ---------------------------------------------------------------------------
// Stage 2: class-sum reduction and final loss.
// loss = (1/N) * sum_c cnt[c] * log((totalS - clsSum[c])/(N - cnt[c]) + 1e-12)
// ---------------------------------------------------------------------------
__global__ __launch_bounds__(TPB) void fourd_finalize(
    const float* __restrict__ S, const int* __restrict__ labx,
    float* __restrict__ out, int N)
{
  __shared__ float csum[256];
  __shared__ int   ccnt[256];
  __shared__ float red[256];
  int t = threadIdx.x;
  csum[t] = 0.0f;
  ccnt[t] = 0;
  __syncthreads();
  for (int j = t; j < N; j += TPB) {
    int c = labx[j] & 255;
    atomicAdd(&csum[c], S[j]);
    atomicAdd(&ccnt[c], 1);
  }
  __syncthreads();
  red[t] = csum[t];
  __syncthreads();
  for (int s = 128; s > 0; s >>= 1) {
    if (t < s) red[t] += red[t + s];
    __syncthreads();
  }
  float totalS = red[0];
  __syncthreads();
  float term = 0.0f;
  int cc = ccnt[t];
  if (cc > 0) {
    float x = (totalS - csum[t]) / (float)(N - cc);
    term = (float)cc * logf(x + 1e-12f);
  }
  red[t] = term;
  __syncthreads();
  for (int s = 128; s > 0; s >>= 1) {
    if (t < s) red[t] += red[t + s];
    __syncthreads();
  }
  if (t == 0) out[0] = red[0] / (float)N;
}

// ---------------------------------------------------------------------------
extern "C" void kernel_launch(void* const* d_in, const int* in_sizes, int n_in,
                              void* d_out, int out_size, void* d_ws, size_t ws_size,
                              hipStream_t stream) {
  const float* features = (const float*)d_in[0];
  const int*   labels   = (const int*)d_in[1];
  int bsz = in_sizes[1];
  int N   = 2 * bsz;                       // n_views = 2 (reference)
  int d   = in_sizes[0] / N;
  int kchunks = d / 32;

  char* ws = (char*)d_ws;
  __bf16* Ahi  = (__bf16*)ws;                                    // N*d bf16
  __bf16* Alo  = (__bf16*)(ws + (size_t)N * d * 2);              // N*d bf16
  float*  Svec = (float*)(ws + (size_t)N * d * 4);               // N f32
  int*    labx = (int*)(ws + (size_t)N * d * 4 + (size_t)N * 4); // N int

  long total = (long)N * d;
  int prepBlocks = (int)((total + TPB - 1) / TPB);
  fourd_prep<<<prepBlocks, TPB, 0, stream>>>(features, labels, Ahi, Alo, labx,
                                             N, d, bsz, kchunks);

  int ctPerGroup = (N / 16) / 8;   // column tiles per wave (8 groups/block)
  fourd_main<<<N / 32, TPB, 0, stream>>>(Ahi, Alo, labx, Svec, N, kchunks, ctPerGroup);

  fourd_finalize<<<1, TPB, 0, stream>>>(Svec, labx, (float*)d_out, N);
}
